// ResidualBlock_19224273617073
// MI455X (gfx1250) — matrix operand order repaired and checked by reading, not compiled, not run on hardware
//
#include <hip/hip_runtime.h>

// ---------------------------------------------------------------------------
// Mamba residual block for MI455X (gfx1250, wave32).
// GEMMs via v_wmma_f32_16x16x32_bf16 (bf16 in, f32 accumulate).
// Selective scan fused (deltaA/deltaB_u/gate computed on the fly).
// ---------------------------------------------------------------------------

#define B_DIM   2
#define L_DIM   2048
#define DM      1024              // d_model
#define DI      2048              // d_inner
#define DCONV   4
#define DTR     64                // dt_rank
#define DST     16                // d_state
#define NXD     (DTR + 2 * DST)   // 96 (x_proj output width)
#define MROWS   (B_DIM * L_DIM)   // 4096

typedef __attribute__((ext_vector_type(16))) __bf16 v16bf;
typedef __attribute__((ext_vector_type(8)))  float  v8f;

union AFrag { v16bf v; uint4 q[2]; };
union CFrag { v8f   v; float f[8]; };

__device__ __forceinline__ unsigned short f32_to_bf16(float f) {
  unsigned int u = __float_as_uint(f);
  u += 0x7fffu + ((u >> 16) & 1u);          // round-to-nearest-even
  return (unsigned short)(u >> 16);
}

// ---------------------------------------------------------------------------
// RMSNorm: one block per row of x (MROWS rows, DM cols) -> bf16 output.
// ---------------------------------------------------------------------------
__global__ __launch_bounds__(256)
void rmsnorm_kernel(const float* __restrict__ x, const float* __restrict__ w,
                    unsigned short* __restrict__ xn_bf) {
  __shared__ float red[256];
  const int row = blockIdx.x;
  const float* xr = x + (size_t)row * DM;
  float ss = 0.f;
  for (int i = threadIdx.x; i < DM; i += 256) { float v = xr[i]; ss += v * v; }
  red[threadIdx.x] = ss;
  __syncthreads();
  for (int s = 128; s > 0; s >>= 1) {
    if (threadIdx.x < s) red[threadIdx.x] += red[threadIdx.x + s];
    __syncthreads();
  }
  const float scale = rsqrtf(red[0] * (1.0f / DM) + 1e-5f);
  for (int i = threadIdx.x; i < DM; i += 256)
    xn_bf[(size_t)row * DM + i] = f32_to_bf16(xr[i] * scale * w[i]);
}

// ---------------------------------------------------------------------------
// Strided f32 -> bf16 conversion (weights once per launch; dt extraction).
// ---------------------------------------------------------------------------
__global__ __launch_bounds__(256)
void cvt_bf16_kernel(const float* __restrict__ src, unsigned short* __restrict__ dst,
                     long long rows, int cols, int src_stride) {
  long long idx = (long long)blockIdx.x * 256 + threadIdx.x;
  if (idx >= rows * (long long)cols) return;
  long long r = idx / cols;
  int c = (int)(idx - r * cols);
  dst[idx] = f32_to_bf16(src[r * (long long)src_stride + c]);
}

// ---------------------------------------------------------------------------
// WMMA bf16 GEMM: C[M,N] = A[M,K] * W[N,K]^T (+ optional f32 residual).
// Block = 256 threads (8 waves); block tile 128x64; wave tile 32x32
// (2x2 of 16x16x32 WMMA micro-tiles). K must be a multiple of 32;
// M a multiple of 128; N bounds-checked (needed for N=96).
// Fragment layouts per CDNA5 ISA 7.12.2 (wave32).
// ---------------------------------------------------------------------------
__global__ __launch_bounds__(256)
void gemm_bf16_wmma(const __bf16* __restrict__ A, const __bf16* __restrict__ W,
                    float* __restrict__ C, const float* __restrict__ residual,
                    int M, int N, int K) {
  const int lane = threadIdx.x & 31;
  const int wave = threadIdx.x >> 5;      // 0..7
  const int wm   = wave >> 1;             // 0..3
  const int wn   = wave & 1;              // 0..1
  const int m_base = blockIdx.y * 128 + wm * 32;
  const int n_base = blockIdx.x * 64  + wn * 32;

  const int lm  = lane & 15;              // M (A/C) or N (B) index within tile
  const int lhi = lane >> 4;              // half-wave selector
  const int a_koff = lhi * 8;             // A: K sub-offset {0,8}
  const int b_koff = lhi * 8;             // B: K sub-offset {0,8}

  CFrag acc[2][2];
  acc[0][0].v = 0.f; acc[0][1].v = 0.f; acc[1][0].v = 0.f; acc[1][1].v = 0.f;

  for (int k0 = 0; k0 < K; k0 += 32) {
    AFrag a[2], b[2];
#pragma unroll
    for (int i = 0; i < 2; ++i) {
      // A 16x32 bf16: elems 0..7 = K[a_koff..a_koff+7], elems 8..15 = +16.
      const int row = m_base + i * 16 + lm;
      const uint4* p = reinterpret_cast<const uint4*>(A + (size_t)row * K + k0 + a_koff);
      a[i].q[0] = p[0];
      a[i].q[1] = p[2];                   // +32 bytes -> K + 16
    }
#pragma unroll
    for (int j = 0; j < 2; ++j) {
      // B 32x16 bf16: elems 0..7 = K[b_koff..b_koff+7], elems 8..15 = +16;
      // W is (N,K) row-major.
      int n = n_base + j * 16 + lm;
      n = (n < N) ? n : (N - 1);          // clamp loads; stores masked below
      const uint4* p = reinterpret_cast<const uint4*>(W + (size_t)n * K + k0 + b_koff);
      b[j].q[0] = p[0];
      b[j].q[1] = p[2];
    }
#pragma unroll
    for (int i = 0; i < 2; ++i)
#pragma unroll
      for (int j = 0; j < 2; ++j)
        acc[i][j].v = __builtin_amdgcn_wmma_f32_16x16x32_bf16(
            false, a[i].v, false, b[j].v, (short)0, acc[i][j].v, false, false);
  }

#pragma unroll
  for (int i = 0; i < 2; ++i) {
#pragma unroll
    for (int j = 0; j < 2; ++j) {
      const int col = n_base + j * 16 + lm;
      if (col < N) {
#pragma unroll
        for (int r = 0; r < 8; ++r) {
          const int row = m_base + i * 16 + lhi * 8 + r;  // C: VGPR r -> M=r / M=8+r
          const size_t idx = (size_t)row * N + col;
          float v = acc[i][j].f[r];
          if (residual) v += residual[idx];
          C[idx] = v;
        }
      }
    }
  }
}

// ---------------------------------------------------------------------------
// Depthwise causal conv1d (width 4) + bias + SiLU on the u-branch of xz.
// Reads column d of the 4096-wide xz rows; writes u (f32) and u (bf16).
// grid = (DI/256, L, B)
// ---------------------------------------------------------------------------
__global__ __launch_bounds__(256)
void conv_silu_kernel(const float* __restrict__ xz, const float* __restrict__ cw,
                      const float* __restrict__ cb, float* __restrict__ u,
                      unsigned short* __restrict__ u_bf) {
  const int d = blockIdx.x * 256 + threadIdx.x;
  const int l = blockIdx.y;
  const int b = blockIdx.z;
  float acc = cb[d];
#pragma unroll
  for (int j = 0; j < DCONV; ++j) {
    const int ls = l - (DCONV - 1) + j;
    if (ls >= 0)
      acc += cw[d * DCONV + j] * xz[((size_t)(b * L_DIM + ls)) * (2 * DI) + d];
  }
  const float s = acc / (1.f + __expf(-acc));  // SiLU
  const size_t o = ((size_t)(b * L_DIM + l)) * DI + d;
  u[o]    = s;
  u_bf[o] = f32_to_bf16(s);
}

// ---------------------------------------------------------------------------
// delta = softplus(dt_proj_out + dt_proj_b[d])  (in place)
// ---------------------------------------------------------------------------
__global__ __launch_bounds__(256)
void softplus_bias_kernel(float* __restrict__ delta, const float* __restrict__ bias,
                          long long total, int cols) {
  long long idx = (long long)blockIdx.x * 256 + threadIdx.x;
  if (idx >= total) return;
  const int d = (int)(idx % cols);
  const float v = delta[idx] + bias[d];
  delta[idx] = (v > 20.f) ? v : log1pf(__expf(v));
}

// ---------------------------------------------------------------------------
// Fused selective scan. One thread per (b,d) channel; 16-wide state in
// registers; B/C vectors staged through LDS in 8-step chunks (shared by the
// 256 channels of the block). Fuses deltaA, deltaB_u, C-reduction, D-skip
// and the SiLU(z) gate; emits y in bf16 for the out_proj WMMA GEMM.
// grid = (DI/256, B)
// ---------------------------------------------------------------------------
__global__ __launch_bounds__(256)
void scan_kernel(const float* __restrict__ delta, const float* __restrict__ u,
                 const float* __restrict__ xz, const float* __restrict__ xdbl,
                 const float* __restrict__ A_log, const float* __restrict__ D_skip,
                 unsigned short* __restrict__ y_bf) {
  __shared__ float Bsh[8][DST];
  __shared__ float Csh[8][DST];
  const int d = blockIdx.x * 256 + threadIdx.x;
  const int b = blockIdx.y;

  float Af[DST], h[DST];
#pragma unroll
  for (int n = 0; n < DST; ++n) {
    Af[n] = -__expf(A_log[d * DST + n]);  // A = -exp(A_log)
    h[n]  = 0.f;
  }
  const float Dk = D_skip[d];

  const int sld = threadIdx.x >> 5;   // 0..7 : step within chunk
  const int q   = threadIdx.x & 31;   // 0..31: B (0..15) / C (16..31) element

  for (int l0 = 0; l0 < L_DIM; l0 += 8) {
    const float v = xdbl[((size_t)(b * L_DIM + l0 + sld)) * NXD + DTR + q];
    __syncthreads();
    if (q < DST) Bsh[sld][q] = v; else Csh[sld][q - DST] = v;
    __syncthreads();

    for (int s = 0; s < 8; ++s) {
      const int l = l0 + s;
      const size_t o = ((size_t)(b * L_DIM + l)) * DI + d;
      const float dlt = delta[o];
      const float uu  = u[o];
      float y = 0.f;
#pragma unroll
      for (int n = 0; n < DST; ++n) {
        const float dA = __expf(dlt * Af[n]);
        h[n] = dA * h[n] + (dlt * Bsh[s][n]) * uu;
        y += h[n] * Csh[s][n];
      }
      y += uu * Dk;
      const float z = xz[((size_t)(b * L_DIM + l)) * (2 * DI) + DI + d];
      y *= z / (1.f + __expf(-z));      // gate: y * silu(z)
      y_bf[o] = f32_to_bf16(y);
    }
  }
}

// ---------------------------------------------------------------------------
extern "C" void kernel_launch(void* const* d_in, const int* in_sizes, int n_in,
                              void* d_out, int out_size, void* d_ws, size_t ws_size,
                              hipStream_t stream) {
  const float* x         = (const float*)d_in[0];
  const float* norm_w    = (const float*)d_in[1];
  const float* in_proj_w = (const float*)d_in[2];   // (2*DI, DM)
  const float* conv_w    = (const float*)d_in[3];   // (DI, 1, 4)
  const float* conv_b    = (const float*)d_in[4];
  const float* x_proj_w  = (const float*)d_in[5];   // (NXD, DI)
  const float* dt_proj_w = (const float*)d_in[6];   // (DI, DTR)
  const float* dt_proj_b = (const float*)d_in[7];
  const float* A_log     = (const float*)d_in[8];   // (DI, DST)
  const float* D_skip    = (const float*)d_in[9];
  const float* out_proj_w= (const float*)d_in[10];  // (DM, DI)
  float* out = (float*)d_out;

  char* ws = (char*)d_ws;
  size_t off = 0;
  auto alloc = [&](size_t bytes) -> void* {
    void* p = ws + off;
    off = (off + bytes + 255) & ~(size_t)255;
    return p;
  };

  unsigned short* xn_bf   = (unsigned short*)alloc((size_t)MROWS * DM * 2);
  unsigned short* w_in_bf = (unsigned short*)alloc((size_t)2 * DI * DM * 2);
  unsigned short* w_xp_bf = (unsigned short*)alloc((size_t)NXD * DI * 2);
  unsigned short* w_dt_bf = (unsigned short*)alloc((size_t)DI * DTR * 2);
  unsigned short* w_out_bf= (unsigned short*)alloc((size_t)DM * DI * 2);
  float*          xz      = (float*)alloc((size_t)MROWS * 2 * DI * 4);
  float*          u       = (float*)alloc((size_t)MROWS * DI * 4);
  unsigned short* u_bf    = (unsigned short*)alloc((size_t)MROWS * DI * 2);
  float*          xdbl    = (float*)alloc((size_t)MROWS * NXD * 4);
  unsigned short* dt_bf   = (unsigned short*)alloc((size_t)MROWS * DTR * 2);
  float*          delta   = (float*)alloc((size_t)MROWS * DI * 4);
  unsigned short* y_bf    = (unsigned short*)alloc((size_t)MROWS * DI * 2);
  (void)ws_size; (void)in_sizes; (void)n_in; (void)out_size;

  // 1. RMSNorm -> bf16 activations
  rmsnorm_kernel<<<MROWS, 256, 0, stream>>>(x, norm_w, xn_bf);

  // 2. Convert weights to bf16
  auto cvt = [&](const float* s, unsigned short* dpt, long long rows, int cols, int stride) {
    long long tot = rows * cols;
    cvt_bf16_kernel<<<dim3((unsigned)((tot + 255) / 256)), 256, 0, stream>>>(s, dpt, rows, cols, stride);
  };
  cvt(in_proj_w,  w_in_bf,  2 * DI, DM,  DM);
  cvt(x_proj_w,   w_xp_bf,  NXD,    DI,  DI);
  cvt(dt_proj_w,  w_dt_bf,  DI,     DTR, DTR);
  cvt(out_proj_w, w_out_bf, DM,     DI,  DI);

  // 3. in_proj GEMM: xz[4096, 4096] = xn[4096,1024] * in_proj_w^T
  gemm_bf16_wmma<<<dim3((2 * DI) / 64, MROWS / 128), 256, 0, stream>>>(
      (const __bf16*)xn_bf, (const __bf16*)w_in_bf, xz, nullptr, MROWS, 2 * DI, DM);

  // 4. depthwise conv + SiLU on u branch
  conv_silu_kernel<<<dim3(DI / 256, L_DIM, B_DIM), 256, 0, stream>>>(
      xz, conv_w, conv_b, u, u_bf);

  // 5. x_proj GEMM: xdbl[4096, 96] = u * x_proj_w^T  (N=96 bounds-checked)
  gemm_bf16_wmma<<<dim3((NXD + 63) / 64, MROWS / 128), 256, 0, stream>>>(
      (const __bf16*)u_bf, (const __bf16*)w_xp_bf, xdbl, nullptr, MROWS, NXD, DI);

  // 6. extract dt columns (0..63) -> bf16
  cvt(xdbl, dt_bf, MROWS, DTR, NXD);

  // 7. dt_proj GEMM: delta_raw[4096, 2048] = dt * dt_proj_w^T
  gemm_bf16_wmma<<<dim3(DI / 64, MROWS / 128), 256, 0, stream>>>(
      (const __bf16*)dt_bf, (const __bf16*)w_dt_bf, delta, nullptr, MROWS, DI, DTR);

  // 8. delta = softplus(delta_raw + bias)
  {
    long long tot = (long long)MROWS * DI;
    softplus_bias_kernel<<<dim3((unsigned)((tot + 255) / 256)), 256, 0, stream>>>(
        delta, dt_proj_b, tot, DI);
  }

  // 9. fused selective scan + D-skip + SiLU gate -> y (bf16)
  scan_kernel<<<dim3(DI / 256, B_DIM), 256, 0, stream>>>(
      delta, u, xz, xdbl, A_log, D_skip, y_bf);

  // 10. out_proj GEMM + residual: out[4096,1024] = y * out_proj_w^T + x
  gemm_bf16_wmma<<<dim3(DM / 64, MROWS / 128), 256, 0, stream>>>(
      (const __bf16*)y_bf, (const __bf16*)w_out_bf, out, x, MROWS, DM, DI);
}
